// DistanceConv2_51196010168704
// MI455X (gfx1250) — matrix operand order repaired and checked by reading, not compiled, not run on hardware
//
#include <hip/hip_runtime.h>
#include <hip/hip_bf16.h>
#include <stdint.h>

// ---------------------------------------------------------------------------
// Edge-MLP: out[e] = ReLU([x[src]|x[dst]|attr] @ W1^T + b1) @ W2^T + b2
// Persistent-weight CDNA5 WMMA kernel, v4:
//   - weights swizzled once to bf16 A-fragment order, pinned in VGPRs
//   - division-free cooperative gather, double-buffered: global loads issued
//     at iteration top (LOADcnt only), LDS stores deferred until after L1
//   - dual accumulator chains in both layers to overlap DS latency with WMMA
// ---------------------------------------------------------------------------

typedef __attribute__((ext_vector_type(16))) __bf16 v16bf;
typedef __attribute__((ext_vector_type(8)))  float  v8f;
typedef __attribute__((ext_vector_type(4)))  float  f32x4;
typedef __attribute__((ext_vector_type(4)))  unsigned int u32x4;
typedef __attribute__((ext_vector_type(2)))  unsigned int u32x2;

static constexpr int E_EDGES = 800000;
static constexpr int IN_CH   = 128;
static constexpr int HOP     = 16;
static constexpr int DIN     = 2 * IN_CH + HOP;   // 272
static constexpr int HID     = 256;
static constexpr int OUTC    = 128;

static constexpr int KT1 = 9;    // ceil(288/32): K padded 272 -> 288
static constexpr int NT1 = 16;
static constexpr int KT2 = 8;
static constexpr int NT2 = 8;

static constexpr int W1A_ELEMS = NT1 * KT1 * 32 * 16;   // 73728 bf16
static constexpr int W2A_ELEMS = NT2 * KT2 * 32 * 16;   // 32768 bf16
static constexpr size_t W2A_OFF = (size_t)W1A_ELEMS * 2;

static constexpr int NBLOCKS = 1000;
static constexpr int NTILES  = E_EDGES / 32;      // 25000 = 1000 * 25 exactly

static constexpr int ESTRIDE = 296;  // bf16 elems; 592 B rows, bank-staggered
static constexpr int HSTRIDE = 264;  // bf16 elems; 528 B rows, bank-staggered

__device__ __host__ inline int afrag_k(int half, int e) {
    return half * 16 + e;
}

__global__ void prep_w1_kernel(const float* __restrict__ W1,
                               __bf16* __restrict__ w1a) {
    int idx = blockIdx.x * 256 + threadIdx.x;       // [nt][kt][lane][e]
    if (idx >= W1A_ELEMS) return;
    int e    = idx & 15;
    int lane = (idx >> 4) & 31;
    int kt   = (idx >> 9) % KT1;
    int nt   = idx / (512 * KT1);
    int m    = lane & 15, half = lane >> 4;
    int k    = kt * 32 + afrag_k(half, e);
    int hid  = nt * 16 + m;
    float v  = (k < DIN) ? W1[hid * DIN + k] : 0.0f;
    w1a[idx] = (__bf16)v;
}

__global__ void prep_w2_kernel(const float* __restrict__ W2,
                               __bf16* __restrict__ w2a) {
    int idx = blockIdx.x * 256 + threadIdx.x;       // [ot][kt][lane][e]
    if (idx >= W2A_ELEMS) return;
    int e    = idx & 15;
    int lane = (idx >> 4) & 31;
    int kt   = (idx >> 9) & 7;
    int ot   = idx >> 12;
    int m    = lane & 15, half = lane >> 4;
    int k    = kt * 32 + afrag_k(half, e);
    int o    = ot * 16 + m;
    w2a[idx] = (__bf16)W2[o * HID + k];
}

__device__ inline unsigned int pack2_raw(float a, float b) {
    unsigned short lo = __builtin_bit_cast(unsigned short, (__bf16)a);
    unsigned short hi = __builtin_bit_cast(unsigned short, (__bf16)b);
    return (unsigned int)lo | ((unsigned int)hi << 16);
}
__device__ inline unsigned int relu_pack2(float a, float b) {
    return pack2_raw(fmaxf(a, 0.0f), fmaxf(b, 0.0f));
}
__device__ inline u32x2 pack_quad(f32x4 f) {
    u32x2 r = { pack2_raw(f[0], f[1]), pack2_raw(f[2], f[3]) };
    return r;
}

union U16BF { u32x4 q[2]; v16bf v; };

__global__ void __launch_bounds__(512)
edge_mlp_wmma_kernel(const float*   __restrict__ x,
                     const int64_t* __restrict__ eidx,
                     const float*   __restrict__ eattr,
                     const float*   __restrict__ b1,
                     const float*   __restrict__ b2,
                     const __bf16*  __restrict__ w1a,
                     const __bf16*  __restrict__ w2a,
                     float*         __restrict__ out) {
    __shared__ __align__(16) unsigned short est[2][32][ESTRIDE]; // 37.9 KB
    __shared__ __align__(16) unsigned short hbuf[2][16][HSTRIDE]; // 16.9 KB

    const int tid  = threadIdx.x;
    const int wave = tid >> 5;
    const int lane = tid & 31;
    const int er   = lane & 15;
    const int half = lane >> 4;

    const int nt  = wave;        // layer-1 hidden tile owned by this wave
    const int ot  = wave & 7;    // layer-2 output tile
    const int et2 = wave >> 3;   // layer-2 edge sub-tile

    // ---- pin weight fragments in VGPRs ----
    const v16bf* w1v = (const v16bf*)w1a;
    const v16bf* w2v = (const v16bf*)w2a;
    v16bf a1f[KT1], a2f[KT2];
#pragma unroll
    for (int kt = 0; kt < KT1; ++kt) a1f[kt] = w1v[(nt * KT1 + kt) * 32 + lane];
#pragma unroll
    for (int kt = 0; kt < KT2; ++kt) a2f[kt] = w2v[(ot * KT2 + kt) * 32 + lane];

    const f32x4* bp1 = (const f32x4*)(b1 + nt * 16 + half * 8);
    const f32x4  b1l = bp1[0], b1h = bp1[1];
    const f32x4* bp2 = (const f32x4*)(b2 + ot * 16 + half * 8);
    const f32x4  b2l = bp2[0], b2h = bp2[1];

    // ---- loop-invariant gather geometry (division-free) ----
    // x-part: 2048 quads = 512 threads x 4; this thread's column is fixed:
    const int grp   = tid >> 6;             // edge sub-row 0..7 (step 8 per i)
    const int kq    = (tid & 63) * 4;       // f32 column 0..252 (fixed)
    const bool useS = (kq < 128);           // src half vs dst half of concat
    const int64_t* eix = useS ? eidx : (eidx + E_EDGES);
    const int kqx   = useS ? kq : (kq - 128);
    const int xoff  = grp * ESTRIDE + kq;   // LDS elem offset, + i*8*ESTRIDE
    // attr part: threads 0..127 (perfectly coalesced); pad: threads 128..255
    const int aoff  = (tid >> 2) * ESTRIDE + 256 + (tid & 3) * 4;
    const int rpad  = tid - 128;
    const int poff  = (rpad >> 2) * ESTRIDE + 272 + (rpad & 3) * 4;

    // ---- prologue: stage first tile directly into est[0] ----
    {
        const long eb0 = (long)blockIdx.x * 32;
        unsigned short* eb = &est[0][0][0];
#pragma unroll
        for (int i = 0; i < 4; ++i) {
            long idxv = (long)eix[eb0 + i * 8 + grp];
            f32x4 f = *(const f32x4*)(x + idxv * IN_CH + kqx);
            *(u32x2*)(eb + xoff + i * 8 * ESTRIDE) = pack_quad(f);
        }
        if (tid < 128) {
            f32x4 f = *(const f32x4*)(eattr + eb0 * HOP + tid * 4);
            *(u32x2*)(eb + aoff) = pack_quad(f);
        } else if (tid < 256) {
            u32x2 z = {0u, 0u};
            *(u32x2*)(eb + poff) = z;
        }
    }

    int parity = 0;
#pragma unroll 1
    for (int t = blockIdx.x; t < NTILES; t += NBLOCKS) {
        __syncthreads();   // est[parity] staged; prev-iter hbuf readers done

        // ---- stage(next): ISSUE global gather loads only (LOADcnt) ----
        const bool hasnext = (t + NBLOCKS) < NTILES;
        const long ebn = (long)(t + NBLOCKS) * 32;
        f32x4 sv[4], sva;
        if (hasnext) {
#pragma unroll
            for (int i = 0; i < 4; ++i) {
                long idxv = (long)eix[ebn + i * 8 + grp];
                sv[i] = *(const f32x4*)(x + idxv * IN_CH + kqx);
            }
            if (tid < 128)
                sva = *(const f32x4*)(eattr + ebn * HOP + tid * 4);
        }

        // ---- layer 1: 18 WMMAs, dual et accumulator chains ----
        {
            const unsigned short* r0 = &est[parity][er][half * 16];
            const unsigned short* r1 = &est[parity][16 + er][half * 16];
            v8f acc0 = {b1l[0], b1l[1], b1l[2], b1l[3],
                        b1h[0], b1h[1], b1h[2], b1h[3]};
            v8f acc1 = acc0;
#pragma unroll
            for (int kt = 0; kt < KT1; ++kt) {
                U16BF c0, c1;
                c0.q[0] = *(const u32x4*)(r0 + kt * 32);
                c0.q[1] = *(const u32x4*)(r0 + kt * 32 + 8);
                c1.q[0] = *(const u32x4*)(r1 + kt * 32);
                c1.q[1] = *(const u32x4*)(r1 + kt * 32 + 8);
                acc0 = __builtin_amdgcn_wmma_f32_16x16x32_bf16(
                    false, a1f[kt], false, c0.v, (short)0, acc0, false, false);
                acc1 = __builtin_amdgcn_wmma_f32_16x16x32_bf16(
                    false, a1f[kt], false, c1.v, (short)0, acc1, false, false);
            }
            u32x4 h0, h1;
            h0[0] = relu_pack2(acc0[0], acc0[1]);  h0[1] = relu_pack2(acc0[2], acc0[3]);
            h0[2] = relu_pack2(acc0[4], acc0[5]);  h0[3] = relu_pack2(acc0[6], acc0[7]);
            h1[0] = relu_pack2(acc1[0], acc1[1]);  h1[1] = relu_pack2(acc1[2], acc1[3]);
            h1[2] = relu_pack2(acc1[4], acc1[5]);  h1[3] = relu_pack2(acc1[6], acc1[7]);
            *(u32x4*)&hbuf[0][er][nt * 16 + half * 8] = h0;  // ds_store_b128
            *(u32x4*)&hbuf[1][er][nt * 16 + half * 8] = h1;
        }

        // ---- stage(next): convert + LDS stores (data arrived during L1) ----
        if (hasnext) {
            unsigned short* eb = &est[parity ^ 1][0][0];
#pragma unroll
            for (int i = 0; i < 4; ++i)
                *(u32x2*)(eb + xoff + i * 8 * ESTRIDE) = pack_quad(sv[i]);
            if (tid < 128) {
                *(u32x2*)(eb + aoff) = pack_quad(sva);
            } else if (tid < 256) {
                u32x2 z = {0u, 0u};
                *(u32x2*)(eb + poff) = z;
            }
        }
        __syncthreads();   // hbuf ready (staging stores also retired here)

        // ---- layer 2: 8 WMMAs, even/odd K split -> dual chains ----
        {
            const unsigned short* hr = &hbuf[et2][er][half * 16];
            v8f accA = {b2l[0], b2l[1], b2l[2], b2l[3],
                        b2h[0], b2h[1], b2h[2], b2h[3]};
            v8f accB = {0.f, 0.f, 0.f, 0.f, 0.f, 0.f, 0.f, 0.f};
#pragma unroll
            for (int kt = 0; kt < KT2; ++kt) {
                U16BF c;
                c.q[0] = *(const u32x4*)(hr + kt * 32);
                c.q[1] = *(const u32x4*)(hr + kt * 32 + 8);
                if (kt & 1)
                    accB = __builtin_amdgcn_wmma_f32_16x16x32_bf16(
                        false, a2f[kt], false, c.v, (short)0, accB, false, false);
                else
                    accA = __builtin_amdgcn_wmma_f32_16x16x32_bf16(
                        false, a2f[kt], false, c.v, (short)0, accA, false, false);
            }
            v8f acc = accA + accB;
            const long e = (long)t * 32 + et2 * 16 + er;
            float* op = out + e * OUTC + ot * 16 + half * 8;
            f32x4 o0 = {acc[0], acc[1], acc[2], acc[3]};
            f32x4 o1 = {acc[4], acc[5], acc[6], acc[7]};
            ((f32x4*)op)[0] = o0;      // 2x global_store_b128
            ((f32x4*)op)[1] = o1;
        }
        parity ^= 1;
    }
}

extern "C" void kernel_launch(void* const* d_in, const int* in_sizes, int n_in,
                              void* d_out, int out_size, void* d_ws, size_t ws_size,
                              hipStream_t stream) {
    const float*   x     = (const float*)d_in[0];
    const int64_t* eidx  = (const int64_t*)d_in[1];
    const float*   eattr = (const float*)d_in[2];
    const float*   W1    = (const float*)d_in[3];
    const float*   b1    = (const float*)d_in[4];
    const float*   W2    = (const float*)d_in[5];
    const float*   b2    = (const float*)d_in[6];
    float*         out   = (float*)d_out;

    __bf16* w1a = (__bf16*)d_ws;
    __bf16* w2a = (__bf16*)((char*)d_ws + W2A_OFF);

    prep_w1_kernel<<<W1A_ELEMS / 256, 256, 0, stream>>>(W1, w1a);
    prep_w2_kernel<<<W2A_ELEMS / 256, 256, 0, stream>>>(W2, w2a);

    edge_mlp_wmma_kernel<<<NBLOCKS, 512, 0, stream>>>(
        x, eidx, eattr, b1, b2, w1a, w2a, out);
}
